// ViT_Generator_47038481825860
// MI455X (gfx1250) — compile-verified, matrix operand order checked by reading, not run on hardware
//
#include <hip/hip_runtime.h>
#include <hip/hip_bf16.h>

// ---------------- problem constants (from reference) ----------------
#define BB     64
#define CC     3
#define HH     256
#define WW     256
#define PP     16
#define DD     1024
#define DEPTH  10
#define HEADS  2
#define DHD    64
#define MLPD   256
#define INNERD 128
#define NN     288         // sequence length  (16*16 patches + 2*16 cls)
#define PDD    768         // patch dim
#define NPATCH 256
#define NROWS  (BB*NN)     // 18432
#define SCL    0.125f      // DH^-0.5

typedef _Float16 h16;
typedef _Float16 v16h __attribute__((ext_vector_type(16)));
typedef float    v8f  __attribute__((ext_vector_type(8)));
typedef unsigned int v4u __attribute__((ext_vector_type(4)));
typedef int      v4i  __attribute__((ext_vector_type(4)));
typedef int      v8i  __attribute__((ext_vector_type(8)));

union AFrag { v16h v; uint4 q[2]; h16 h[16]; };
union CFrag { v8f  v; float f[8]; };

static __device__ inline v8f wmma_f16(v16h a, v16h b, v8f c) {
    return __builtin_amdgcn_wmma_f32_16x16x32_f16(false, a, false, b, (short)0, c, false, false);
}

// ---------------- Tensor Data Mover staging (gfx1250) ----------------
#if defined(__has_builtin)
#if __has_builtin(__builtin_amdgcn_tensor_load_to_lds) && __has_builtin(__builtin_amdgcn_s_wait_tensorcnt)
#define HAVE_TDM 1
#endif
#endif
#ifndef HAVE_TDM
#define HAVE_TDM 0
#endif

#if HAVE_TDM
static __device__ inline unsigned lds_off_of(const void* p) {
    return (unsigned)(unsigned long long)(__attribute__((address_space(3))) const void*)p;
}
// 2D tile load of 2-byte elements: tile (tile_w x tile_h) from a row-major
// tensor with row stride `stride_elems`; OOB (>= tensor_w / tensor_h,
// expressed as dims REMAINING from the tile origin) reads return zero.
// Optional LDS row padding: pad_amount+1 DWORDs inserted every 8<<pad_interval bytes.
static __device__ inline void tdm_load_tile_2d(
    unsigned lds_byte_off, const h16* gtile,
    unsigned tile_w, unsigned tile_h,
    unsigned tensor_w, unsigned tensor_h,
    unsigned stride_elems,
    unsigned pad_enable, unsigned pad_interval, unsigned pad_amount)
{
    unsigned long long ga = (unsigned long long)gtile;
    v4u g0 = { 1u,                                   // count=1 (valid descriptor)
               lds_byte_off,                         // lds_addr
               (unsigned)ga,                         // global_addr[31:0]
               (unsigned)((ga >> 32) & 0x01ffffffu) | (2u << 30) }; // addr[56:32] | type=2
    v8i g1;
    g1[0] = (int)((1u << 16)                         // data_size = 1 (2 bytes)
                | (pad_enable << 20)
                | (pad_interval << 22)
                | (pad_amount << 25));
    g1[1] = (int)((tensor_w & 0xffffu) << 16);       // tensor_dim0[15:0] at bits[63:48]
    g1[2] = (int)((tensor_w >> 16) | ((tensor_h & 0xffffu) << 16)); // dim0 hi | dim1 lo
    g1[3] = (int)((tensor_h >> 16) | (tile_w << 16));               // dim1 hi | tile_dim0
    g1[4] = (int)(tile_h & 0xffffu);                 // tile_dim1 (tile_dim2 = 0)
    g1[5] = (int)stride_elems;                       // tensor_dim0_stride[31:0]
    g1[6] = 0;                                       // stride hi / dim1_stride lo
    g1[7] = 0;
    v4i gz = {0, 0, 0, 0};
#if defined(__clang_major__) && (__clang_major__ >= 23)
    v8i gz8 = {0, 0, 0, 0, 0, 0, 0, 0};
    __builtin_amdgcn_tensor_load_to_lds(g0, g1, gz, gz, gz8, 0);
#else
    __builtin_amdgcn_tensor_load_to_lds(g0, g1, gz, gz, 0);
#endif
}
#endif // HAVE_TDM

// ---------------- fp32 -> fp16 convert ----------------
__global__ void f32_to_f16(const float* __restrict__ in, h16* __restrict__ o, long n) {
    long i = (long)blockIdx.x * blockDim.x + threadIdx.x;
    if (i < n) o[i] = (h16)in[i];
}

// ---------------- im2col of image patches (fp16) ----------------
__global__ void im2col(const float* __restrict__ img, h16* __restrict__ Xp, long n) {
    long id = (long)blockIdx.x * blockDim.x + threadIdx.x;
    if (id >= n) return;
    int d  = (int)(id % PDD);
    long t = id / PDD;
    int np = (int)(t % NPATCH);
    int b  = (int)(t / NPATCH);
    int hp = np >> 4, wp = np & 15;
    int p  = d / 48;
    int q  = (d % 48) / 3;
    int c  = d % 3;
    size_t src = (((size_t)b*CC + c)*HH + (hp*16 + p))*WW + (wp*16 + q);
    Xp[id] = (h16)img[src];
}

// ---------------- generic WMMA GEMM ----------------
// tile 128(M) x 64(N), 8 waves (4x2), each wave 32x32 (2x2 wmma frags), k-step 32.
// A tile is DMA'd by the Tensor Data Mover (with LDS row padding to a 48-half
// pitch); B tile is staged+transposed by the waves and overlaps the TDM fill.
template<bool GELU, bool RES, bool OUT32, bool OUT16>
__global__ __launch_bounds__(256)
void gemm_wmma(const h16* __restrict__ A, const h16* __restrict__ Bm,
               const float* __restrict__ bias, const float* __restrict__ resid,
               float* __restrict__ out32, h16* __restrict__ out16,
               int M, int Nn, int K, int row_block, int row_stride, int row_base)
{
    __shared__ h16 sA[128][48];   // [row][k]   pitch 48 halves (96B, 16B aligned)
    __shared__ h16 sB[64][48];    // [col][k]   (B tile stored transposed)

    const int tid  = threadIdx.x;
    const int lane = tid & 31;
    const int wave = tid >> 5;
    const int wm   = wave >> 1;        // 0..3
    const int wn   = wave & 1;         // 0..1
    const int m0   = blockIdx.y * 128;
    const int n0   = blockIdx.x * 64;

    CFrag c[2][2];
#pragma unroll
    for (int fi = 0; fi < 2; ++fi)
#pragma unroll
        for (int fj = 0; fj < 2; ++fj)
#pragma unroll
            for (int e = 0; e < 8; ++e) c[fi][fj].f[e] = 0.0f;

    const int hi = (lane >= 16) ? 8 : 0;

    for (int k0 = 0; k0 < K; k0 += 32) {
#if HAVE_TDM
        // ---- A tile via TDM: 32x128 tile, rows padded 16->24 DWORDs in LDS ----
        if (wave == 0) {
            tdm_load_tile_2d(lds_off_of(&sA[0][0]),
                             A + (size_t)m0 * K + k0,
                             /*tile_w*/32, /*tile_h*/128,
                             /*tensor_w*/(unsigned)(K - k0),
                             /*tensor_h*/(unsigned)(M - m0),
                             /*stride*/(unsigned)K,
                             /*pad_en*/1u, /*pad_interval(16dw)*/3u, /*pad_amount(8dw)*/7u);
        }
#else
#pragma unroll
        for (int i = 0; i < 2; ++i) {
            int chunk = tid * 2 + i;          // 0..511
            int r = chunk >> 2, part = chunk & 3;
            int gr = m0 + r;
            uint4 val{0u,0u,0u,0u};
            if (gr < M) val = *(const uint4*)(A + (size_t)gr * K + k0 + part * 8);
            *(uint4*)&sA[r][part * 8] = val;
        }
        if (k0 + 32 < K) {
            int gr = m0 + (tid >> 1);
            if (gr < M) __builtin_prefetch(A + (size_t)gr * K + k0 + 32, 0, 1);
        }
#endif
        // ---- stage B tile 32x64, transposed into sB[col][k] (overlaps TDM) ----
        {
            int kr = tid >> 3, cp = tid & 7;  // 32 k-rows x 8 col-groups
            uint4 val = *(const uint4*)(Bm + (size_t)(k0 + kr) * Nn + n0 + cp * 8);
            const h16* hv = (const h16*)&val;
#pragma unroll
            for (int e = 0; e < 8; ++e) sB[cp * 8 + e][kr] = hv[e];
        }
#if HAVE_TDM
        if (wave == 0) __builtin_amdgcn_s_wait_tensorcnt(0);
#endif
        __syncthreads();

        // ---- fragments + 4 WMMAs ----
        AFrag a[2], b[2];
#pragma unroll
        for (int fi = 0; fi < 2; ++fi) {
            int r = wm * 32 + fi * 16 + (lane & 15);
            a[fi].q[0] = *(const uint4*)&sA[r][hi];
            a[fi].q[1] = *(const uint4*)&sA[r][16 + hi];
        }
#pragma unroll
        for (int fj = 0; fj < 2; ++fj) {
            int cc = wn * 32 + fj * 16 + (lane & 15);
            b[fj].q[0] = *(const uint4*)&sB[cc][hi];
            b[fj].q[1] = *(const uint4*)&sB[cc][16 + hi];
        }
#pragma unroll
        for (int fi = 0; fi < 2; ++fi)
#pragma unroll
            for (int fj = 0; fj < 2; ++fj)
                c[fi][fj].v = wmma_f16(a[fi].v, b[fj].v, c[fi][fj].v);
        __syncthreads();
    }

    // ---- epilogue ----
#pragma unroll
    for (int fi = 0; fi < 2; ++fi) {
        int rbase = m0 + wm * 32 + fi * 16 + ((lane >= 16) ? 8 : 0);
#pragma unroll
        for (int fj = 0; fj < 2; ++fj) {
            int col = n0 + wn * 32 + fj * 16 + (lane & 15);
#pragma unroll
            for (int e = 0; e < 8; ++e) {
                int gr = rbase + e;
                if (gr < M) {
                    float val = c[fi][fj].f[e];
                    if (bias) val += bias[col];
                    if (GELU) val = 0.5f * val * (1.0f + erff(val * 0.70710678118f));
                    size_t orow = (size_t)row_base + (size_t)(gr / row_block) * row_stride
                                + (size_t)(gr % row_block);
                    size_t off = orow * (size_t)Nn + col;
                    if (RES)   val += resid[off];
                    if (OUT32) out32[off] = val;
                    if (OUT16) out16[off] = (h16)val;
                }
            }
        }
    }
}

// ---------------- assemble sequence: cls rows + pos_emb ----------------
__global__ void assemble(float* __restrict__ x, const float* __restrict__ cls_emb,
                         const float* __restrict__ pos, long n) {
    long id = (long)blockIdx.x * blockDim.x + threadIdx.x;
    if (id >= n) return;
    int d  = (int)(id % DD);
    long t = id / DD;
    int nn = (int)(t % NN);
    float v;
    if (nn < 16)        v = cls_emb[(size_t)nn * DD + d];
    else if (nn >= 272) v = cls_emb[(size_t)(nn - 256) * DD + d];
    else                v = x[id];
    x[id] = v + pos[(size_t)nn * DD + d];
}

// ---------------- LayerNorm -> fp16 ----------------
__global__ __launch_bounds__(256)
void ln_kernel(const float* __restrict__ x, const float* __restrict__ s,
               const float* __restrict__ b, h16* __restrict__ y)
{
    __shared__ float wsum[8], wsq[8];
    int row = blockIdx.x;
    int tid = threadIdx.x, lane = tid & 31, wave = tid >> 5;
    const float* xr = x + (size_t)row * DD;
    float4 v = ((const float4*)xr)[tid];
    float sum = v.x + v.y + v.z + v.w;
    float sq  = v.x*v.x + v.y*v.y + v.z*v.z + v.w*v.w;
#pragma unroll
    for (int o = 16; o > 0; o >>= 1) {
        sum += __shfl_xor(sum, o, 32);
        sq  += __shfl_xor(sq,  o, 32);
    }
    if (lane == 0) { wsum[wave] = sum; wsq[wave] = sq; }
    __syncthreads();
    float tsum = 0.f, tsq = 0.f;
#pragma unroll
    for (int i = 0; i < 8; ++i) { tsum += wsum[i]; tsq += wsq[i]; }
    float mean = tsum * (1.0f / DD);
    float var  = tsq * (1.0f / DD) - mean * mean;
    float rinv = rsqrtf(var + 1e-5f);
    float vv[4] = {v.x, v.y, v.z, v.w};
#pragma unroll
    for (int k = 0; k < 4; ++k) {
        int j = tid * 4 + k;
        y[(size_t)row * DD + j] = (h16)((vv[k] - mean) * rinv * s[j] + b[j]);
    }
}

// ---------------- attention: softmax(Q K^T * scale) -> fp16 probs ----------------
__global__ __launch_bounds__(128)
void attn_scores(const h16* __restrict__ qkv, h16* __restrict__ Ah)
{
    __shared__ h16  sK[NN][64];
    __shared__ h16  sQ[16][64];
    __shared__ float sS[16][NN];
    __shared__ float red[16][8];

    int bid = blockIdx.x;
    int mt  = bid % 18;
    int bh  = bid / 18;
    int h   = bh % HEADS;
    int b   = bh / HEADS;
    int tid = threadIdx.x, lane = tid & 31, wave = tid >> 5;
    const h16* base = qkv + ((size_t)b * NN) * 384 + h * 64;

#if HAVE_TDM
    if (wave == 0) {   // K block: 64-wide x 288 rows of the strided qkv tensor
        tdm_load_tile_2d(lds_off_of(&sK[0][0]), base + 128,
                         /*tile_w*/64, /*tile_h*/NN,
                         /*tensor_w*/64, /*tensor_h*/NN, /*stride*/384,
                         0u, 0u, 0u);
    }
#else
    for (int i = tid; i < NN * 8; i += 128) {
        int j = i >> 3, part = i & 7;
        *(uint4*)&sK[j][part * 8] = *(const uint4*)(base + (size_t)j * 384 + 128 + part * 8);
    }
#endif
    for (int i = tid; i < 16 * 8; i += 128) {       // Q rows for this tile
        int r = i >> 3, part = i & 7;
        *(uint4*)&sQ[r][part * 8] = *(const uint4*)(base + (size_t)(mt * 16 + r) * 384 + part * 8);
    }
#if HAVE_TDM
    if (wave == 0) __builtin_amdgcn_s_wait_tensorcnt(0);
#endif
    __syncthreads();

    const int hi = (lane >= 16) ? 8 : 0;
    AFrag aq[2];
#pragma unroll
    for (int ks = 0; ks < 2; ++ks) {
        int r = lane & 15;
        aq[ks].q[0] = *(const uint4*)&sQ[r][ks * 32 + hi];
        aq[ks].q[1] = *(const uint4*)&sQ[r][ks * 32 + 16 + hi];
    }
    for (int jt = wave; jt < 18; jt += 4) {
        CFrag acc;
#pragma unroll
        for (int e = 0; e < 8; ++e) acc.f[e] = 0.0f;
#pragma unroll
        for (int ks = 0; ks < 2; ++ks) {
            AFrag bk;
            int jc = jt * 16 + (lane & 15);
            bk.q[0] = *(const uint4*)&sK[jc][ks * 32 + hi];
            bk.q[1] = *(const uint4*)&sK[jc][ks * 32 + 16 + hi];
            acc.v = wmma_f16(aq[ks].v, bk.v, acc.v);
        }
        int r0 = (lane >= 16) ? 8 : 0;
#pragma unroll
        for (int e = 0; e < 8; ++e)
            sS[r0 + e][jt * 16 + (lane & 15)] = acc.f[e] * SCL;
    }
    __syncthreads();

    // softmax over 288 per row: 16 rows x 8 threads
    int r = tid >> 3, g = tid & 7;
    float mx = -3.4e38f;
    for (int j = g; j < NN; j += 8) mx = fmaxf(mx, sS[r][j]);
    red[r][g] = mx; __syncthreads();
    if (g == 0) { float m = red[r][0]; for (int i = 1; i < 8; ++i) m = fmaxf(m, red[r][i]); red[r][0] = m; }
    __syncthreads();
    mx = red[r][0];
    float sum = 0.f;
    for (int j = g; j < NN; j += 8) { float e = __expf(sS[r][j] - mx); sS[r][j] = e; sum += e; }
    __syncthreads();
    red[r][g] = sum; __syncthreads();
    if (g == 0) { float m = 0.f; for (int i = 0; i < 8; ++i) m += red[r][i]; red[r][0] = m; }
    __syncthreads();
    float inv = 1.0f / red[r][0];
    h16* dst = Ah + ((size_t)bh * NN + mt * 16 + r) * NN;
    for (int j = g; j < NN; j += 8) dst[j] = (h16)(sS[r][j] * inv);
}

// ---------------- attention: O = probs @ V -> fp16 [row, h*64+d] ----------------
__global__ __launch_bounds__(128)
void attn_out(const h16* __restrict__ Ah, const h16* __restrict__ qkv, h16* __restrict__ Oh)
{
    __shared__ h16 sA[16][NN];
    __shared__ h16 sV[64][NN];   // [d][j]

    int bid = blockIdx.x;
    int mt  = bid % 18;
    int bh  = bid / 18;
    int h   = bh % HEADS;
    int b   = bh / HEADS;
    int tid = threadIdx.x, lane = tid & 31, wave = tid >> 5;
    const h16* base = qkv + ((size_t)b * NN) * 384 + h * 64;

#if HAVE_TDM
    if (wave == 0) {   // probs tile: contiguous 288 x 16
        tdm_load_tile_2d(lds_off_of(&sA[0][0]),
                         Ah + ((size_t)bh * NN + mt * 16) * NN,
                         /*tile_w*/NN, /*tile_h*/16,
                         /*tensor_w*/NN, /*tensor_h*/16, /*stride*/NN,
                         0u, 0u, 0u);
    }
#else
    for (int i = tid; i < 16 * 36; i += 128) {
        int r = i / 36, part = i % 36;
        *(uint4*)&sA[r][part * 8] =
            *(const uint4*)(Ah + ((size_t)bh * NN + mt * 16 + r) * NN + part * 8);
    }
#endif
    for (int i = tid; i < NN * 8; i += 128) {   // V (col offset 256), transposed
        int j = i >> 3, part = i & 7;
        uint4 val = *(const uint4*)(base + (size_t)j * 384 + 256 + part * 8);
        const h16* hv = (const h16*)&val;
#pragma unroll
        for (int e = 0; e < 8; ++e) sV[part * 8 + e][j] = hv[e];
    }
#if HAVE_TDM
    if (wave == 0) __builtin_amdgcn_s_wait_tensorcnt(0);
#endif
    __syncthreads();

    const int hi = (lane >= 16) ? 8 : 0;
    CFrag acc;
#pragma unroll
    for (int e = 0; e < 8; ++e) acc.f[e] = 0.0f;
    int dc = wave * 16 + (lane & 15);
#pragma unroll
    for (int ks = 0; ks < 9; ++ks) {
        AFrag a, bv;
        int r = lane & 15;
        a.q[0]  = *(const uint4*)&sA[r][ks * 32 + hi];
        a.q[1]  = *(const uint4*)&sA[r][ks * 32 + 16 + hi];
        bv.q[0] = *(const uint4*)&sV[dc][ks * 32 + hi];
        bv.q[1] = *(const uint4*)&sV[dc][ks * 32 + 16 + hi];
        acc.v = wmma_f16(a.v, bv.v, acc.v);
    }
    int r0 = (lane >= 16) ? 8 : 0;
#pragma unroll
    for (int e = 0; e < 8; ++e)
        Oh[((size_t)b * NN + mt * 16 + r0 + e) * INNERD + h * 64 + wave * 16 + (lane & 15)]
            = (h16)acc.f[e];
}

// ---------------- generator: normalized-dot argmax per batch ----------------
__global__ __launch_bounds__(256)
void gen_argmax(const float* __restrict__ g, int* __restrict__ idx)
{
    __shared__ float srow[PDD];
    __shared__ float sval[256];
    __shared__ int   sidx[256];
    int b = blockIdx.x, tid = threadIdx.x;

    int jj = 16 + tid, wp = jj >> 4, hp = jj & 15;
    int n  = hp * 18 + wp;
    const float* crow = g + ((size_t)b * NN + n) * PDD;
    float nrm = 0.f;
    for (int d = 0; d < PDD; ++d) { float t = crow[d]; nrm += t * t; }
    nrm = fmaxf(sqrtf(nrm), 1e-12f);

    for (int i = 0; i < 32; ++i) {
        int js = (i < 16) ? i : (272 + (i - 16));
        int wps = js >> 4, hps = js & 15;
        int ns = hps * 18 + wps;
        const float* srcrow = g + ((size_t)b * NN + ns) * PDD;
        for (int d = tid; d < PDD; d += 256) srow[d] = srcrow[d];
        __syncthreads();
        float dot = 0.f;
        for (int d = 0; d < PDD; ++d) dot += srow[d] * crow[d];
        sval[tid] = dot / nrm; sidx[tid] = tid;
        __syncthreads();
        for (int off = 128; off > 0; off >>= 1) {
            if (tid < off) {
                float v2 = sval[tid + off]; int i2 = sidx[tid + off];
                if (v2 > sval[tid] || (v2 == sval[tid] && i2 < sidx[tid])) {
                    sval[tid] = v2; sidx[tid] = i2;
                }
            }
            __syncthreads();
        }
        if (tid == 0) idx[b * 32 + i] = sidx[0];
        __syncthreads();
    }
}

// ---------------- final layout scatter ----------------
__global__ void final_out(const float* __restrict__ g, const int* __restrict__ idx,
                          float* __restrict__ out, long total)
{
    long id = (long)blockIdx.x * blockDim.x + threadIdx.x;
    if (id >= total) return;
    int x = (int)(id % 288);  long t = id / 288;
    int y = (int)(t % 256);   t /= 256;
    int c = (int)(t % 3);
    int b = (int)(t / 3);
    int wp = x >> 4, q = x & 15;
    int hp = y >> 4, p = y & 15;
    int j = wp * 16 + hp;                       // g_final row
    int jj;
    if (j < 16)        jj = 16 + idx[b * 32 + j];
    else if (j >= 272) jj = 16 + idx[b * 32 + 16 + (j - 272)];
    else               jj = j;
    int wp2 = jj >> 4, hp2 = jj & 15;
    int n = hp2 * 18 + wp2;
    int d = (p * 16 + q) * 3 + c;
    out[id] = g[((size_t)b * NN + n) * PDD + d];
}

// ==================== host launcher ====================
extern "C" void kernel_launch(void* const* d_in, const int* in_sizes, int n_in,
                              void* d_out, int out_size, void* d_ws, size_t ws_size,
                              hipStream_t stream)
{
    (void)in_sizes; (void)n_in; (void)out_size; (void)ws_size;
    const float* img   = (const float*)d_in[0];
    const float* W_pe  = (const float*)d_in[1];
    const float* b_pe  = (const float*)d_in[2];
    const float* cls_l = (const float*)d_in[3];
    const float* cls_r = (const float*)d_in[4];
    const float* W_lin = (const float*)d_in[5];
    const float* b_lin = (const float*)d_in[6];
    const float* pos   = (const float*)d_in[7];
    const float* ln1_s = (const float*)d_in[8];
    const float* ln1_b = (const float*)d_in[9];
    const float* Wqkv  = (const float*)d_in[10];
    const float* Wo    = (const float*)d_in[11];
    const float* bo    = (const float*)d_in[12];
    const float* ln2_s = (const float*)d_in[13];
    const float* ln2_b = (const float*)d_in[14];
    const float* W1    = (const float*)d_in[15];
    const float* b1    = (const float*)d_in[16];
    const float* W2    = (const float*)d_in[17];
    const float* b2    = (const float*)d_in[18];
    const float* lng_s = (const float*)d_in[19];
    const float* lng_b = (const float*)d_in[20];
    const float* Wg    = (const float*)d_in[21];
    const float* bg    = (const float*)d_in[22];
    float* out = (float*)d_out;

    char* ws = (char*)d_ws;
    size_t off = 0;
    auto alloc = [&](size_t bytes) -> void* {
        void* p = ws + off;
        off += (bytes + 255) & ~(size_t)255;
        return p;
    };
    h16*  W_pe_h  = (h16*) alloc((size_t)PDD * DD * 2);
    h16*  W_lin_h = (h16*) alloc((size_t)PDD * DD * 2);
    h16*  Wg_h    = (h16*) alloc((size_t)DD * PDD * 2);
    h16*  Wqkv_h  = (h16*) alloc((size_t)DEPTH * DD * 384 * 2);
    h16*  Wo_h    = (h16*) alloc((size_t)DEPTH * INNERD * DD * 2);
    h16*  W1_h    = (h16*) alloc((size_t)DEPTH * DD * MLPD * 2);
    h16*  W2_h    = (h16*) alloc((size_t)DEPTH * MLPD * DD * 2);
    h16*  cls_h   = (h16*) alloc((size_t)32 * PDD * 2);
    float* clsemb = (float*)alloc((size_t)32 * DD * 4);
    h16*  Xp      = (h16*) alloc((size_t)BB * NPATCH * PDD * 2);
    float* xbuf   = (float*)alloc((size_t)NROWS * DD * 4);
    h16*  yh      = (h16*) alloc((size_t)NROWS * DD * 2);
    h16*  qkvh    = (h16*) alloc((size_t)NROWS * 384 * 2);
    h16*  Ahb     = (h16*) alloc((size_t)BB * HEADS * NN * NN * 2);
    h16*  Oh      = (h16*) alloc((size_t)NROWS * INNERD * 2);
    h16*  hh      = (h16*) alloc((size_t)NROWS * MLPD * 2);
    float* gbuf   = (float*)alloc((size_t)NROWS * PDD * 4);
    int*  idxb    = (int*)  alloc((size_t)BB * 32 * 4);

    auto cvt = [&](const float* src, h16* dst, long n) {
        f32_to_f16<<<(unsigned)((n + 255) / 256), 256, 0, stream>>>(src, dst, n);
    };
    cvt(W_pe,  W_pe_h,  (long)PDD * DD);
    cvt(W_lin, W_lin_h, (long)PDD * DD);
    cvt(Wg,    Wg_h,    (long)DD * PDD);
    cvt(Wqkv,  Wqkv_h,  (long)DEPTH * DD * 384);
    cvt(Wo,    Wo_h,    (long)DEPTH * INNERD * DD);
    cvt(W1,    W1_h,    (long)DEPTH * DD * MLPD);
    cvt(W2,    W2_h,    (long)DEPTH * MLPD * DD);
    cvt(cls_l, cls_h,               (long)16 * PDD);
    cvt(cls_r, cls_h + 16 * PDD,    (long)16 * PDD);

    {   // im2col
        long n = (long)BB * NPATCH * PDD;
        im2col<<<(unsigned)((n + 255) / 256), 256, 0, stream>>>(img, Xp, n);
    }

    const int BIG = 1 << 30;
    auto grid = [](int M, int Nn) { return dim3((unsigned)(Nn / 64), (unsigned)((M + 127) / 128)); };

    // cls tokens: [32,768] @ W_lin -> clsemb (fp32, + b_lin)
    gemm_wmma<false,false,true,false><<<grid(32, DD), 256, 0, stream>>>(
        cls_h, W_lin_h, b_lin, nullptr, clsemb, nullptr, 32, DD, PDD, BIG, 0, 0);
    // patches: [B*256,768] @ W_pe -> xbuf rows 16..271 per batch (+ b_pe)
    gemm_wmma<false,false,true,false><<<grid(BB*NPATCH, DD), 256, 0, stream>>>(
        Xp, W_pe_h, b_pe, nullptr, xbuf, nullptr, BB*NPATCH, DD, PDD, NPATCH, NN, 16);
    {   // cls rows + pos_emb
        long n = (long)NROWS * DD;
        assemble<<<(unsigned)((n + 255) / 256), 256, 0, stream>>>(xbuf, clsemb, pos, n);
    }

    const unsigned attn_blocks = BB * HEADS * 18;   // 2304
    for (int i = 0; i < DEPTH; ++i) {
        ln_kernel<<<NROWS, 256, 0, stream>>>(xbuf, ln1_s + (size_t)i*DD, ln1_b + (size_t)i*DD, yh);
        gemm_wmma<false,false,false,true><<<grid(NROWS, 384), 256, 0, stream>>>(
            yh, Wqkv_h + (size_t)i*DD*384, nullptr, nullptr, nullptr, qkvh,
            NROWS, 384, DD, BIG, 0, 0);
        attn_scores<<<attn_blocks, 128, 0, stream>>>(qkvh, Ahb);
        attn_out   <<<attn_blocks, 128, 0, stream>>>(Ahb, qkvh, Oh);
        gemm_wmma<false,true,true,false><<<grid(NROWS, DD), 256, 0, stream>>>(
            Oh, Wo_h + (size_t)i*INNERD*DD, bo + (size_t)i*DD, xbuf, xbuf, nullptr,
            NROWS, DD, INNERD, BIG, 0, 0);
        ln_kernel<<<NROWS, 256, 0, stream>>>(xbuf, ln2_s + (size_t)i*DD, ln2_b + (size_t)i*DD, yh);
        gemm_wmma<true,false,false,true><<<grid(NROWS, MLPD), 256, 0, stream>>>(
            yh, W1_h + (size_t)i*DD*MLPD, b1 + (size_t)i*MLPD, nullptr, nullptr, hh,
            NROWS, MLPD, DD, BIG, 0, 0);
        gemm_wmma<false,true,true,false><<<grid(NROWS, DD), 256, 0, stream>>>(
            hh, W2_h + (size_t)i*MLPD*DD, b2 + (size_t)i*DD, xbuf, xbuf, nullptr,
            NROWS, DD, MLPD, BIG, 0, 0);
    }

    ln_kernel<<<NROWS, 256, 0, stream>>>(xbuf, lng_s, lng_b, yh);
    gemm_wmma<false,false,true,false><<<grid(NROWS, PDD), 256, 0, stream>>>(
        yh, Wg_h, bg, nullptr, gbuf, nullptr, NROWS, PDD, DD, BIG, 0, 0);

    gen_argmax<<<BB, 256, 0, stream>>>(gbuf, idxb);
    {
        long total = (long)BB * CC * 256 * 288;
        final_out<<<(unsigned)((total + 255) / 256), 256, 0, stream>>>(gbuf, idxb, out, total);
    }
}
